// GCNLayer_32650341384775
// MI455X (gfx1250) — compile-verified, matrix operand order checked by reading, not run on hardware
//
#include <hip/hip_runtime.h>
#include <hip/hip_bf16.h>

// ---------------------------------------------------------------------------
// Types for CDNA5 WMMA bf16: D(16x16 f32) = A(16x32 bf16) x B(32x16 bf16) + C
// ---------------------------------------------------------------------------
typedef __attribute__((ext_vector_type(16))) __bf16 bf16x16;
typedef __attribute__((ext_vector_type(8)))  __bf16 bf16x8;
typedef __attribute__((ext_vector_type(8)))  float  f32x8;

union BFragU { bf16x16 v; bf16x8 h[2]; };

#define HID   256
#define MT    32    // edge-kernel M tile (rows per workgroup)
#define SAS   520   // bf16 LDS row stride (256/512 cols + pad) -> bank spread
#define SFS   260   // f32 LDS row stride

__device__ __forceinline__ __bf16 f2bf(float f) {
    union { float f; unsigned u; } in; in.f = f;
    unsigned u = in.u;
    unsigned r = u + 0x7FFFu + ((u >> 16) & 1u);       // round-to-nearest-even
    union { unsigned short s; __bf16 b; } o;
    o.s = (unsigned short)(r >> 16);
    return o.b;
}

// gfx1250 async global->LDS staging (ASYNCcnt-tracked DMA path)
__device__ __forceinline__ void async_b128(unsigned lds_byte_off, const float* gsrc) {
    asm volatile("global_load_async_to_lds_b128 %0, %1, off"
                 :: "v"(lds_byte_off), "v"(gsrc) : "memory");
}
__device__ __forceinline__ void wait_async0() {
    asm volatile("s_wait_asynccnt 0" ::: "memory");
}

// One 16x16 output tile: A from LDS (row-major bf16, stride sas), B from global
// bf16 W^T laid out [n][k]. ISA 7.12.2 16-bit layouts: lane<16 -> K {0..7,16..23},
// lane>=16 -> +8; K swept in chunks of 32.
__device__ __forceinline__ f32x8 wmma_rowtile(const __bf16* sA, int sas,
                                              const __bf16* WT, int dk,
                                              int n0, int lane) {
    f32x8 acc = {0.f, 0.f, 0.f, 0.f, 0.f, 0.f, 0.f, 0.f};
    const int mrow  = lane & 15;
    const int khalf = (lane >> 4) & 1;
    const int n     = n0 + (lane & 15);
    const __bf16* arow = sA + (size_t)mrow * sas;
    const __bf16* brow = WT + (size_t)n * dk;
    for (int c = 0; c < dk; c += 32) {
        const int kb = c + khalf * 8;
        BFragU a, b;
        a.h[0] = *(const bf16x8*)(arow + kb);
        a.h[1] = *(const bf16x8*)(arow + kb + 16);
        b.h[0] = *(const bf16x8*)(brow + kb);
        b.h[1] = *(const bf16x8*)(brow + kb + 16);
        acc = __builtin_amdgcn_wmma_f32_16x16x32_bf16(
            false, a.v, false, b.v, (short)0, acc, false, false);
    }
    return acc;
}

// Two 16x16 tiles (rows 0-15 and 16-31 of a 32-row M tile) sharing one set of
// B-fragment loads -> halves L2 weight traffic per output row.
__device__ __forceinline__ void wmma_rowtile2(const __bf16* sA, int sas,
                                              const __bf16* WT, int dk,
                                              int n0, int lane,
                                              f32x8& acc0, f32x8& acc1) {
    const int mrow  = lane & 15;
    const int khalf = (lane >> 4) & 1;
    const int n     = n0 + (lane & 15);
    const __bf16* a0 = sA + (size_t)mrow * sas;
    const __bf16* a1 = sA + (size_t)(mrow + 16) * sas;
    const __bf16* brow = WT + (size_t)n * dk;
    for (int c = 0; c < dk; c += 32) {
        const int kb = c + khalf * 8;
        BFragU b, x0, x1;
        b.h[0]  = *(const bf16x8*)(brow + kb);
        b.h[1]  = *(const bf16x8*)(brow + kb + 16);
        x0.h[0] = *(const bf16x8*)(a0 + kb);
        x0.h[1] = *(const bf16x8*)(a0 + kb + 16);
        x1.h[0] = *(const bf16x8*)(a1 + kb);
        x1.h[1] = *(const bf16x8*)(a1 + kb + 16);
        acc0 = __builtin_amdgcn_wmma_f32_16x16x32_bf16(
            false, x0.v, false, b.v, (short)0, acc0, false, false);
        acc1 = __builtin_amdgcn_wmma_f32_16x16x32_bf16(
            false, x1.v, false, b.v, (short)0, acc1, false, false);
    }
}

// ---------------------------------------------------------------------------
// K0: convert weight W[k][n] (f32) -> WT[n][k] (bf16): B-fragments become
// contiguous 16B global loads per lane.
// ---------------------------------------------------------------------------
__global__ __launch_bounds__(256) void wconv_kernel(const float* __restrict__ W,
                                                    __bf16* __restrict__ WT,
                                                    int dk, int dn) {
    int idx = blockIdx.x * 256 + threadIdx.x;
    if (idx >= dk * dn) return;
    int n = idx / dk, k = idx - n * dk;
    WT[idx] = f2bf(W[(size_t)k * dn + n]);
}

// ---------------------------------------------------------------------------
// K1: generic 16-row M-tile GEMM: out[r][0:256] = A[r][0:dk] @ W + bias.
// 256 threads = 8 waves, each wave owns 2 of the 16 N tiles. Optional
// neighbor-gather for A rows (attn_k / attn_v path).
// ---------------------------------------------------------------------------
__global__ __launch_bounds__(256) void gemm16_kernel(
    const float* __restrict__ A, int lda, int dk,
    const __bf16* __restrict__ WT, const float* __restrict__ bias,
    float* __restrict__ out, int ostride, int ooffs,
    const int* __restrict__ gidx, int gdiv, const float* __restrict__ gsrc) {
    __shared__ __bf16 sA[16 * SAS];
    const int tid = threadIdx.x;
    const int m0  = blockIdx.x * 16;
    for (int idx = tid; idx < 16 * dk; idx += 256) {
        int m = idx / dk, c = idx - m * dk;
        int row = m0 + m;
        const float* src;
        if (gidx) {
            int b = row / gdiv;
            src = gsrc + ((size_t)(b * HID + gidx[row])) * HID;
        } else {
            src = A + (size_t)row * lda;
        }
        sA[m * SAS + c] = f2bf(src[c]);
    }
    __syncthreads();
    const int wave = tid >> 5, lane = tid & 31;
    const int khalf = (lane >> 4) & 1;
    for (int t = 0; t < 2; ++t) {
        const int n0 = (wave * 2 + t) * 16;
        const int n  = n0 + (lane & 15);
        f32x8 acc = wmma_rowtile(sA, SAS, WT, dk, n0, lane);
        for (int r = 0; r < 8; ++r) {
            int m = r + khalf * 8;
            out[(size_t)(m0 + m) * ostride + ooffs + n] = acc[r] + bias[n];
        }
    }
}

// ---------------------------------------------------------------------------
// K2: attention over K=16 neighbors. One wave per (b,n) row.
// ---------------------------------------------------------------------------
__global__ __launch_bounds__(256) void attn_kernel(
    const float* __restrict__ q, const float* __restrict__ kbuf,
    const float* __restrict__ vbuf, float* __restrict__ hagg) {
    const int tid = threadIdx.x, wave = tid >> 5, lane = tid & 31;
    const int row = blockIdx.x * 8 + wave;
    const float* qr = q + (size_t)row * HID;
    float sc[16];
    for (int kk = 0; kk < 16; ++kk) {
        const float* kr = kbuf + ((size_t)row * 16 + kk) * HID;
        float p = 0.f;
        for (int c = lane; c < HID; c += 32) p += qr[c] * kr[c];
        for (int o = 16; o; o >>= 1) p += __shfl_xor(p, o, 32);
        sc[kk] = p * (1.0f / 16.0f);            // 1/sqrt(256)
    }
    float mx = sc[0];
    for (int kk = 1; kk < 16; ++kk) mx = fmaxf(mx, sc[kk]);
    float se = 0.f;
    for (int kk = 0; kk < 16; ++kk) { sc[kk] = __expf(sc[kk] - mx); se += sc[kk]; }
    const float inv = 1.0f / se;
    for (int c = lane; c < HID; c += 32) {
        float acc = 0.f;
        for (int kk = 0; kk < 16; ++kk)
            acc += sc[kk] * vbuf[((size_t)row * 16 + kk) * HID + c];
        hagg[(size_t)row * HID + c] = acc * inv;
    }
}

// ---------------------------------------------------------------------------
// K3: out = LayerNorm(a + relu(g)) per 256-wide row. One wave per row.
// ---------------------------------------------------------------------------
__global__ __launch_bounds__(256) void resid_relu_ln_kernel(
    const float* __restrict__ a, int astride,
    const float* __restrict__ g, int gstride,
    const float* __restrict__ gam, const float* __restrict__ bet,
    float* __restrict__ out, int ostride) {
    const int tid = threadIdx.x, wave = tid >> 5, lane = tid & 31;
    const int row = blockIdx.x * 8 + wave;
    const float* ar = a + (size_t)row * astride;
    const float* gr = g + (size_t)row * gstride;
    float x[8], s = 0.f, s2 = 0.f;
    for (int e = 0; e < 8; ++e) {
        int c = lane + e * 32;
        float v = gr[c]; v = v > 0.f ? v : 0.f;
        float y = ar[c] + v;
        x[e] = y; s += y; s2 += y * y;
    }
    for (int o = 16; o; o >>= 1) { s += __shfl_xor(s, o, 32); s2 += __shfl_xor(s2, o, 32); }
    float mu = s * (1.0f / 256.0f);
    float var = s2 * (1.0f / 256.0f) - mu * mu;
    float rs = rsqrtf(var + 1e-5f);
    float* orow = out + (size_t)row * ostride;
    for (int e = 0; e < 8; ++e) {
        int c = lane + e * 32;
        orow[c] = (x[e] - mu) * rs * gam[c] + bet[c];
    }
}

// ---------------------------------------------------------------------------
// K4: fused edge pipeline. Block = 32 edge rows (fixed b,i; 32-wide j tile).
// h_e async-DMA'd to LDS once, h_e_next written once; 4 WMMA GEMM stages with
// B-fragment reuse across the two 16-row M subtiles.
// ---------------------------------------------------------------------------
__global__ __launch_bounds__(256) void edge_fused_kernel(
    const float* __restrict__ he, const float* __restrict__ hfrom,
    const float* __restrict__ hto,
    const __bf16* __restrict__ e1T, const float* __restrict__ b1,
    const __bf16* __restrict__ ecT, const float* __restrict__ bc,
    const __bf16* __restrict__ eaT, const float* __restrict__ ba,
    const __bf16* __restrict__ veT, const float* __restrict__ be,
    const float* __restrict__ gEA, const float* __restrict__ bEA,
    const float* __restrict__ gEC, const float* __restrict__ bEC,
    float* __restrict__ out) {
    __shared__ __bf16 sA[MT * SAS];   // GEMM A operand (bf16), up to 512 wide
    __shared__ float  sE[MT * SFS];   // h_e rows (f32) for residual
    __shared__ float  sT[MT * SFS];   // t1 -> agg -> aggLN
    __shared__ float  sU[MT * SFS];   // comlin -> pre-final-LN
    const int tid = threadIdx.x, wave = tid >> 5, lane = tid & 31;
    const int bid = blockIdx.x;
    const int jt = bid & 7, i = (bid >> 3) & 255, b = bid >> 11;
    const size_t rowbase = ((size_t)(b * HID + i)) * HID + jt * MT; // flat edge row
    const float* heRows = he + rowbase * HID;   // MT*HID contiguous floats

    // 1. async-DMA the f32 tile into LDS (ASYNCcnt path), then bf16-convert
    for (int it = 0; it < 8; ++it) {
        const int f = (it * 256 + tid) * 4;        // flat float index
        const int m = f >> 8, c = f & 255;
        unsigned loff = (unsigned)(size_t)(sE + m * SFS + c);
        async_b128(loff, heRows + f);
    }
    wait_async0();
    __syncthreads();
    for (int idx = tid; idx < MT * HID; idx += 256) {
        int m = idx >> 8, c = idx & 255;
        sA[m * SAS + c] = f2bf(sE[m * SFS + c]);
    }
    __syncthreads();

    const float* fr = hfrom + ((size_t)(b * HID + i)) * HID;
    const float* toBase = hto + ((size_t)(b * HID) + jt * MT) * HID;
    const int khalf = (lane >> 4) & 1;
    const int nl = lane & 15;

    // 2. t1 = e@W1 + b1 + from + to ;  comlin = e@Vec + bc
    for (int t = 0; t < 2; ++t) {
        const int n0 = (wave * 2 + t) * 16, n = n0 + nl;
        f32x8 a0 = {0.f,0.f,0.f,0.f,0.f,0.f,0.f,0.f}, a1 = a0;
        wmma_rowtile2(sA, SAS, e1T, HID, n0, lane, a0, a1);
        f32x8 c0 = {0.f,0.f,0.f,0.f,0.f,0.f,0.f,0.f}, c1 = c0;
        wmma_rowtile2(sA, SAS, ecT, HID, n0, lane, c0, c1);
        for (int r = 0; r < 8; ++r) {
            int m = r + khalf * 8;
            sT[m * SFS + n] = a0[r] + b1[n] + fr[n] + toBase[(size_t)m * HID + n];
            sT[(m + 16) * SFS + n] = a1[r] + b1[n] + fr[n] + toBase[(size_t)(m + 16) * HID + n];
            sU[m * SFS + n] = c0[r] + bc[n];
            sU[(m + 16) * SFS + n] = c1[r] + bc[n];
        }
    }
    __syncthreads();

    // 3. sA <- bf16(t1)
    for (int idx = tid; idx < MT * HID; idx += 256) {
        int m = idx >> 8, c = idx & 255;
        sA[m * SAS + c] = f2bf(sT[m * SFS + c]);
    }
    __syncthreads();

    // 4. agg = e + relu(t1@Wa + ba)
    for (int t = 0; t < 2; ++t) {
        const int n0 = (wave * 2 + t) * 16, n = n0 + nl;
        f32x8 a0 = {0.f,0.f,0.f,0.f,0.f,0.f,0.f,0.f}, a1 = a0;
        wmma_rowtile2(sA, SAS, eaT, HID, n0, lane, a0, a1);
        for (int r = 0; r < 8; ++r) {
            int m = r + khalf * 8;
            float v0 = a0[r] + ba[n];
            float v1 = a1[r] + ba[n];
            sT[m * SFS + n] = sE[m * SFS + n] + (v0 > 0.f ? v0 : 0.f);
            sT[(m + 16) * SFS + n] = sE[(m + 16) * SFS + n] + (v1 > 0.f ? v1 : 0.f);
        }
    }
    __syncthreads();

    // 5. aggLN = LN(agg); build cat = [bf16(comlin) | bf16(aggLN)] in sA
    for (int rr = 0; rr < 4; ++rr) {
        const int m = wave * 4 + rr;
        float s = 0.f, s2 = 0.f;
        for (int c = lane; c < HID; c += 32) { float x = sT[m * SFS + c]; s += x; s2 += x * x; }
        for (int o = 16; o; o >>= 1) { s += __shfl_xor(s, o, 32); s2 += __shfl_xor(s2, o, 32); }
        float mu = s * (1.0f / 256.0f);
        float var = s2 * (1.0f / 256.0f) - mu * mu;
        float rs = rsqrtf(var + 1e-5f);
        for (int c = lane; c < HID; c += 32) {
            float y = (sT[m * SFS + c] - mu) * rs * gEA[c] + bEA[c];
            sT[m * SFS + c] = y;                     // keep f32 aggLN for residual
            sA[m * SAS + 256 + c] = f2bf(y);         // cat upper half
            sA[m * SAS + c] = f2bf(sU[m * SFS + c]); // cat lower half = comlin
        }
    }
    __syncthreads();

    // 6. pre = aggLN + relu(cat @ V_edge + be)   (dk = 512)
    for (int t = 0; t < 2; ++t) {
        const int n0 = (wave * 2 + t) * 16, n = n0 + nl;
        f32x8 a0 = {0.f,0.f,0.f,0.f,0.f,0.f,0.f,0.f}, a1 = a0;
        wmma_rowtile2(sA, SAS, veT, 2 * HID, n0, lane, a0, a1);
        for (int r = 0; r < 8; ++r) {
            int m = r + khalf * 8;
            float v0 = a0[r] + be[n];
            float v1 = a1[r] + be[n];
            sU[m * SFS + n] = sT[m * SFS + n] + (v0 > 0.f ? v0 : 0.f);
            sU[(m + 16) * SFS + n] = sT[(m + 16) * SFS + n] + (v1 > 0.f ? v1 : 0.f);
        }
    }
    __syncthreads();

    // 7. h_e_next = LN(pre) -> global
    for (int rr = 0; rr < 4; ++rr) {
        const int m = wave * 4 + rr;
        float s = 0.f, s2 = 0.f;
        for (int c = lane; c < HID; c += 32) { float x = sU[m * SFS + c]; s += x; s2 += x * x; }
        for (int o = 16; o; o >>= 1) { s += __shfl_xor(s, o, 32); s2 += __shfl_xor(s2, o, 32); }
        float mu = s * (1.0f / 256.0f);
        float var = s2 * (1.0f / 256.0f) - mu * mu;
        float rs = rsqrtf(var + 1e-5f);
        float* orow = out + (rowbase + m) * HID;
        for (int c = lane; c < HID; c += 32)
            orow[c] = (sU[m * SFS + c] - mu) * rs * gEC[c] + bEC[c];
    }
}

// ---------------------------------------------------------------------------
// Host orchestration
// ---------------------------------------------------------------------------
extern "C" void kernel_launch(void* const* d_in, const int* in_sizes, int n_in,
                              void* d_out, int out_size, void* d_ws, size_t ws_size,
                              hipStream_t stream) {
    const int B = 2, N = 256, K = 16, d = HID;
    const int BN = B * N;            // 512
    const int BNK = B * N * K;       // 8192

    // Inputs (setup_inputs dict order, nested dicts flattened w-then-b):
    const float* h_n = (const float*)d_in[0];
    const float* h_e = (const float*)d_in[1];
    const int* nidx  = (const int*)d_in[2];
    const float* Wq  = (const float*)d_in[3];  const float* bq  = (const float*)d_in[4];
    const float* Wk  = (const float*)d_in[5];  const float* bk  = (const float*)d_in[6];
    const float* Wv  = (const float*)d_in[7];  const float* bv  = (const float*)d_in[8];
    const float* Wna = (const float*)d_in[9];  const float* bna = (const float*)d_in[10];
    const float* Wvc = (const float*)d_in[11]; const float* bvc = (const float*)d_in[12];
    const float* Wvn = (const float*)d_in[13]; const float* bvn = (const float*)d_in[14];
    const float* Wea = (const float*)d_in[15]; const float* bea = (const float*)d_in[16];
    const float* We1 = (const float*)d_in[17]; const float* be1 = (const float*)d_in[18];
    const float* We2 = (const float*)d_in[19]; const float* be2 = (const float*)d_in[20];
    const float* We3 = (const float*)d_in[21]; const float* be3 = (const float*)d_in[22];
    const float* Wec = (const float*)d_in[23]; const float* bec = (const float*)d_in[24];
    const float* Wve = (const float*)d_in[25]; const float* bve = (const float*)d_in[26];
    const float* g_na = (const float*)d_in[27]; const float* b_na = (const float*)d_in[28];
    const float* g_nc = (const float*)d_in[29]; const float* b_nc = (const float*)d_in[30];
    const float* g_ea = (const float*)d_in[31]; const float* b_ea = (const float*)d_in[32];
    const float* g_ec = (const float*)d_in[33]; const float* b_ec = (const float*)d_in[34];

    // Workspace carve-up
    char* base = (char*)d_ws;
    size_t off = 0;
    auto carve = [&](size_t bytes) -> char* {
        char* p = base + off;
        off = (off + bytes + 255) & ~(size_t)255;
        return p;
    };
    const size_t wtB  = (size_t)d * d * sizeof(__bf16);      // 128KB
    const size_t wt2B = (size_t)2 * d * d * sizeof(__bf16);  // 256KB
    __bf16* qT  = (__bf16*)carve(wtB);
    __bf16* kT  = (__bf16*)carve(wtB);
    __bf16* vT  = (__bf16*)carve(wtB);
    __bf16* naT = (__bf16*)carve(wtB);
    __bf16* vcT = (__bf16*)carve(wtB);
    __bf16* eaT = (__bf16*)carve(wtB);
    __bf16* e1T = (__bf16*)carve(wtB);
    __bf16* e2T = (__bf16*)carve(wtB);
    __bf16* e3T = (__bf16*)carve(wtB);
    __bf16* ecT = (__bf16*)carve(wtB);
    __bf16* vnT = (__bf16*)carve(wt2B);
    __bf16* veT = (__bf16*)carve(wt2B);
    float* qbuf  = (float*)carve((size_t)BN * d * 4);
    float* hfrom = (float*)carve((size_t)BN * d * 4);
    float* hto   = (float*)carve((size_t)BN * d * 4);
    float* kbuf  = (float*)carve((size_t)BNK * d * 4);
    float* vbuf  = (float*)carve((size_t)BNK * d * 4);
    float* hagg  = (float*)carve((size_t)BN * d * 4);
    float* g1    = (float*)carve((size_t)BN * d * 4);
    float* g2    = (float*)carve((size_t)BN * d * 4);
    float* cat   = (float*)carve((size_t)BN * 2 * d * 4);   // [vcom | aggLN]
    (void)ws_size; (void)n_in; (void)in_sizes; (void)out_size;

    float* out_n = (float*)d_out;                   // (B,N,d)
    float* out_e = (float*)d_out + (size_t)BN * d;  // (B,N,N,d)

    // K0: weight transpose+convert to bf16 [n][k]
    const int wg1 = (d * d) / 256, wg2 = (2 * d * d) / 256;
    wconv_kernel<<<wg1, 256, 0, stream>>>(Wq,  qT,  d, d);
    wconv_kernel<<<wg1, 256, 0, stream>>>(Wk,  kT,  d, d);
    wconv_kernel<<<wg1, 256, 0, stream>>>(Wv,  vT,  d, d);
    wconv_kernel<<<wg1, 256, 0, stream>>>(Wna, naT, d, d);
    wconv_kernel<<<wg1, 256, 0, stream>>>(Wvc, vcT, d, d);
    wconv_kernel<<<wg1, 256, 0, stream>>>(Wea, eaT, d, d);
    wconv_kernel<<<wg1, 256, 0, stream>>>(We1, e1T, d, d);
    wconv_kernel<<<wg1, 256, 0, stream>>>(We2, e2T, d, d);
    wconv_kernel<<<wg1, 256, 0, stream>>>(We3, e3T, d, d);
    wconv_kernel<<<wg1, 256, 0, stream>>>(Wec, ecT, d, d);
    wconv_kernel<<<wg2, 256, 0, stream>>>(Wvn, vnT, 2 * d, d);
    wconv_kernel<<<wg2, 256, 0, stream>>>(Wve, veT, 2 * d, d);

    // Node-side GEMMs (M = 512 -> 32 blocks)
    gemm16_kernel<<<BN / 16, 256, 0, stream>>>(h_n, d, d, qT,  bq,  qbuf,  d, 0, nullptr, 0, nullptr);
    gemm16_kernel<<<BN / 16, 256, 0, stream>>>(h_n, d, d, e2T, be2, hfrom, d, 0, nullptr, 0, nullptr);
    gemm16_kernel<<<BN / 16, 256, 0, stream>>>(h_n, d, d, e3T, be3, hto,   d, 0, nullptr, 0, nullptr);
    gemm16_kernel<<<BN / 16, 256, 0, stream>>>(h_n, d, d, vcT, bvc, cat, 2 * d, 0, nullptr, 0, nullptr);
    // Gathered neighbor GEMMs (M = 8192 -> 512 blocks)
    gemm16_kernel<<<BNK / 16, 256, 0, stream>>>(nullptr, 0, d, kT, bk, kbuf, d, 0, nidx, N * K, h_n);
    gemm16_kernel<<<BNK / 16, 256, 0, stream>>>(nullptr, 0, d, vT, bv, vbuf, d, 0, nidx, N * K, h_n);

    // Attention over K neighbors
    attn_kernel<<<BN / 8, 256, 0, stream>>>(qbuf, kbuf, vbuf, hagg);

    // Node aggregation: g1 = hagg@Wna + b ; aggLN = LN(h_n + relu(g1)) -> cat[:,256:]
    gemm16_kernel<<<BN / 16, 256, 0, stream>>>(hagg, d, d, naT, bna, g1, d, 0, nullptr, 0, nullptr);
    resid_relu_ln_kernel<<<BN / 8, 256, 0, stream>>>(h_n, d, g1, d, g_na, b_na, cat + d, 2 * d);

    // Node combine: g2 = cat@Wvn + b ; h_n_next = LN(aggLN + relu(g2))
    gemm16_kernel<<<BN / 16, 256, 0, stream>>>(cat, 2 * d, 2 * d, vnT, bvn, g2, d, 0, nullptr, 0, nullptr);
    resid_relu_ln_kernel<<<BN / 8, 256, 0, stream>>>(cat + d, 2 * d, g2, d, g_nc, b_nc, out_n, d);

    // Fused edge pipeline: B*N*(N/32) = 4096 blocks
    edge_fused_kernel<<<B * N * (N / MT), 256, 0, stream>>>(
        h_e, hfrom, hto,
        e1T, be1, ecT, bec, eaT, bea, veT, bve,
        g_ea, b_ea, g_ec, b_ec, out_e);
}
